// ScaledDotProductAttention_22909355557453
// MI455X (gfx1250) — matrix operand order repaired and checked by reading, not compiled, not run on hardware
//
#include <hip/hip_runtime.h>

// CDNA5 / gfx1250 fused attention with WMMA + Tensor Data Mover:
//   scores = (Q/sqrt(64)) @ K^T  via v_wmma_f32_16x16x32_f16 (f32 accum)
//   K and V tiles staged into LDS by TDM (tensor_load_to_lds), double-buffered
//   masked softmax staged in LDS (16 x 2048 f32); weights streamed NT to global
//   context = P @ V via WMMA.
// One block = one (b,h) head x 16 queries; 4 waves (wave32), 128 threads.

typedef __attribute__((ext_vector_type(16))) _Float16     v16h;
typedef __attribute__((ext_vector_type(8)))  float        v8f;
typedef __attribute__((ext_vector_type(4)))  unsigned int u32x4;
typedef __attribute__((ext_vector_type(8)))  int          i32x8;
typedef __attribute__((ext_vector_type(4)))  int          i32x4;

#define S_LEN 2048
#define DK    64
#define NW    4                 // waves per block
#define QT    16                // query rows per block
#define KT    16                // key cols per WMMA tile
#define NKT   (S_LEN / KT)      // 128 key tiles

// dynamic-LDS layout (float offsets)
#define OFF_SQ      (QT * S_LEN)                 // scores: 32768 floats
#define OFF_RMAX    (OFF_SQ + QT * DK)           // Q tile: 1024
#define OFF_RSUM    (OFF_RMAX + QT * 8)
#define OFF_KBUF    (OFF_RSUM + QT * 8)          // K staging: NW*2*16*64
#define OFF_VBUF    (OFF_KBUF + NW * 2 * QT * DK)// V staging: NW*2*32*16
#define SMEM_FLOATS (OFF_VBUF + NW * 2 * 32 * KT)

// 2-D TDM load: tile (tile_d1 rows x tile_d0 f32 cols) from a row-major
// tensor with row length/stride tensor_d0, into LDS at byte offset lds_off.
// Descriptor per CDNA5 ISA 8.3/8.4 (count=1, type=2, data_size=4B).
__device__ __forceinline__ void tdm_load_2d(unsigned int lds_off,
                                            const float* gptr,
                                            unsigned int tensor_d0,
                                            unsigned int tile_d0,
                                            unsigned int tile_d1)
{
    unsigned long long ga = (unsigned long long)(const void*)gptr;
    u32x4 g0;
    g0[0] = 1u;                                        // count=1, user desc
    g0[1] = lds_off;                                   // lds_addr (bytes)
    g0[2] = (unsigned int)ga;                          // global_addr[31:0]
    g0[3] = (unsigned int)((ga >> 32) & 0x1FFFFFFu)    // global_addr[56:32]
          | (2u << 30);                                // type=2 ("image")
    i32x8 g1;
    g1[0] = 0x20000;                  // data_size=2 (4 bytes); no multicast
    g1[1] = (int)(tensor_d0 << 16);   // tensor_dim0 (low 16 -> bits 63:48)
    g1[2] = (int)(S_LEN << 16);       // tensor_dim1 (low 16 -> bits 95:80)
    g1[3] = (int)(tile_d0 << 16);     // tile_dim0   (bits 127:112)
    g1[4] = (int)tile_d1;             // tile_dim1   (bits 143:128)
    g1[5] = (int)tensor_d0;           // tensor_dim0_stride (bits 191:160)
    g1[6] = 0;
    g1[7] = 0;
    i32x4 z4 = {0, 0, 0, 0};          // groups 2/3 unused (2-D tensor)
    i32x8 z8 = {0, 0, 0, 0, 0, 0, 0, 0};
    __builtin_amdgcn_tensor_load_to_lds(g0, g1, z4, z4, z8, 0);
}

__global__ void __launch_bounds__(128, 1)
sdpa_wmma_kernel(const float* __restrict__ Q,
                 const float* __restrict__ K,
                 const float* __restrict__ V,
                 const unsigned char* __restrict__ mask,  // jnp.bool_ -> 1 byte
                 float* __restrict__ ctx,
                 float* __restrict__ wts)
{
    extern __shared__ float smem[];
    float* sS    = smem;            // [QT][S_LEN]  scores -> probs
    float* sQ    = smem + OFF_SQ;   // [QT][DK]     scaled Q tile
    float* sRmax = smem + OFF_RMAX; // [QT][8]
    float* sRsum = smem + OFF_RSUM; // [QT][8]

    const int tid  = threadIdx.x;
    const int lane = tid & 31;
    const int ln   = lane & 15;     // row for A gathers / col for B,C,D
    const int half = lane >> 4;
    const int w    = __builtin_amdgcn_readfirstlane(tid >> 5);  // wave id (SGPR)

    const int bid = blockIdx.x;     // B*H*(S/16) = 4096 blocks
    const int qt  = bid & (NKT - 1);
    const int bh  = bid >> 7;
    const int q0  = qt * QT;

    const size_t qkvBase  = (size_t)bh * S_LEN * DK;
    const size_t maskBase = (size_t)bh * S_LEN * S_LEN + (size_t)q0 * S_LEN;
    const size_t cBase    = qkvBase + (size_t)q0 * DK;

    // ---- stage 0: Q tile -> LDS, scale 1/sqrt(64) folded in ----
    {
        const float4* src = (const float4*)(Q + qkvBase + (size_t)q0 * DK);
        float4*       dst = (float4*)sQ;
        #pragma unroll
        for (int i = tid; i < QT * DK / 4; i += 128) {
            float4 v = src[i];
            v.x *= 0.125f; v.y *= 0.125f; v.z *= 0.125f; v.w *= 0.125f;
            dst[i] = v;
        }
    }
    __syncthreads();

    // Per-wave A fragments for Q (two 32-wide k-chunks covering DK=64).
    // A 16x32 gather, this code's contiguous k order: lane = M; VGPR v holds K pair at
    // kb = half*16 + v*2.
    v16h aq0, aq1;
    {
        const float2* qp = (const float2*)(sQ + ln * DK);
        #pragma unroll
        for (int v = 0; v < 8; ++v) {
            int p = (half << 3) + v;  // float2 index
            float2 x0 = qp[p];
            float2 x1 = qp[16 + p];
            aq0[2*v]   = (_Float16)x0.x;  aq0[2*v+1] = (_Float16)x0.y;
            aq1[2*v]   = (_Float16)x1.x;  aq1[2*v+1] = (_Float16)x1.y;
        }
    }

    // ---- stage 1: scores = Qs @ K^T; K tiles via TDM double-buffer ----
    // Per-wave staging: two 16x64 f32 buffers.
    tdm_load_2d((OFF_KBUF + (w * 2 + 0) * QT * DK) * 4u,
                K + qkvBase + (size_t)(w * KT) * DK, DK, DK, QT);
    int par = 0;
    for (int t = w; t < NKT; t += NW) {
        const int nt = t + NW;
        if (nt < NKT) {
            tdm_load_2d((OFF_KBUF + (w * 2 + (par ^ 1)) * QT * DK) * 4u,
                        K + qkvBase + (size_t)(nt * KT) * DK, DK, DK, QT);
            __builtin_amdgcn_s_wait_tensorcnt((short)1);   // current buf ready
        } else {
            __builtin_amdgcn_s_wait_tensorcnt((short)0);
        }
        const float* sK = smem + OFF_KBUF + (w * 2 + par) * QT * DK;

        // B (32x16) fragment: lane(ln) = column n = key row; contraction =
        // head dim. Per-column striping: k = half*16 + 2v.
        v16h b0, b1;
        {
            const float2* kp = (const float2*)(sK + ln * DK);
            #pragma unroll
            for (int v = 0; v < 8; ++v) {
                int p = (half << 3) + v;
                float2 x0 = kp[p];
                float2 x1 = kp[16 + p];
                b0[2*v]   = (_Float16)x0.x;  b0[2*v+1] = (_Float16)x0.y;
                b1[2*v]   = (_Float16)x1.x;  b1[2*v+1] = (_Float16)x1.y;
            }
        }
        v8f c = {};
        c = __builtin_amdgcn_wmma_f32_16x16x32_f16(false, aq0, false, b0,
                                                   (short)0, c, false, false);
        c = __builtin_amdgcn_wmma_f32_16x16x32_f16(false, aq1, false, b1,
                                                   (short)0, c, false, false);
        // C/D layout: M = r + 8*half, N = ln. Apply mask at store (NT load:
        // mask is a 128MB single-use stream; keep it out of L2's QKV set).
        const int k0 = t * KT;
        #pragma unroll
        for (int r = 0; r < 8; ++r) {
            int m = r + (half << 3);
            unsigned char mk =
                __builtin_nontemporal_load(mask + maskBase + (size_t)m * S_LEN + k0 + ln);
            sS[m * S_LEN + k0 + ln] = mk ? -1.0e9f : c[r];
        }
        par ^= 1;
    }
    __syncthreads();

    // ---- stage 2: softmax over S in LDS; stream weights to global (NT) ----
    const int rrow = tid >> 3;     // 16 rows x 8 threads each
    const int seg  = tid & 7;
    float* rowp = sS + rrow * S_LEN;
    float mx = -3.0e38f;
    for (int j = seg; j < S_LEN; j += 8) mx = fmaxf(mx, rowp[j]);
    sRmax[rrow * 8 + seg] = mx;
    __syncthreads();
    float rowmax = sRmax[rrow * 8 + 0];
    #pragma unroll
    for (int j = 1; j < 8; ++j) rowmax = fmaxf(rowmax, sRmax[rrow * 8 + j]);
    float sm = 0.0f;
    for (int j = seg; j < S_LEN; j += 8) {
        float e = __expf(rowp[j] - rowmax);
        rowp[j] = e;
        sm += e;
    }
    sRsum[rrow * 8 + seg] = sm;
    __syncthreads();
    float tot = 0.0f;
    #pragma unroll
    for (int j = 0; j < 8; ++j) tot += sRsum[rrow * 8 + j];
    const float inv = 1.0f / tot;
    float* wrow = wts + maskBase + (size_t)rrow * S_LEN;
    for (int j = seg; j < S_LEN; j += 8) {
        float p = rowp[j] * inv;
        rowp[j] = p;                              // normalized P stays for P@V
        __builtin_nontemporal_store(p, wrow + j); // 512MB stream: NT, no reuse
    }
    __syncthreads();

    // ---- stage 3: context = P(16xS) @ V(SxDK); V slices via TDM ----
    // Wave owns a 16-wide slice of D; V tile = 32 rows x 16 cols, 2KB.
    const int n0 = w * KT;
    tdm_load_2d((OFF_VBUF + (w * 2 + 0) * 32 * KT) * 4u,
                V + qkvBase + n0, DK, KT, 32);
    par = 0;
    v8f acc = {};
    for (int kc = 0; kc < S_LEN; kc += 32) {
        const int nk = kc + 32;
        if (nk < S_LEN) {
            tdm_load_2d((OFF_VBUF + (w * 2 + (par ^ 1)) * 32 * KT) * 4u,
                        V + qkvBase + (size_t)nk * DK + n0, DK, KT, 32);
            __builtin_amdgcn_s_wait_tensorcnt((short)1);
        } else {
            __builtin_amdgcn_s_wait_tensorcnt((short)0);
        }
        const float* sV = smem + OFF_VBUF + (w * 2 + par) * 32 * KT;

        // A fragment from P in LDS (same layout as Q gather)
        v16h pa;
        {
            const float2* pp = (const float2*)(sS + ln * S_LEN + kc);
            #pragma unroll
            for (int v = 0; v < 8; ++v) {
                int p = (half << 3) + v;
                float2 x = pp[p];
                pa[2*v] = (_Float16)x.x;  pa[2*v+1] = (_Float16)x.y;
            }
        }
        // B fragment from staged V tile (row-major 32x16 in LDS)
        v16h vb;
        #pragma unroll
        for (int v = 0; v < 8; ++v) {
            int kk = (half << 4) + (v << 1);
            vb[2*v]   = (_Float16)sV[kk * KT + ln];
            vb[2*v+1] = (_Float16)sV[(kk + 1) * KT + ln];
        }
        acc = __builtin_amdgcn_wmma_f32_16x16x32_f16(false, pa, false, vb,
                                                     (short)0, acc, false, false);
        par ^= 1;
    }
    #pragma unroll
    for (int r = 0; r < 8; ++r) {
        int m = r + (half << 3);
        ctx[cBase + (size_t)m * DK + n0 + ln] = acc[r];
    }
}

extern "C" void kernel_launch(void* const* d_in, const int* in_sizes, int n_in,
                              void* d_out, int out_size, void* d_ws, size_t ws_size,
                              hipStream_t stream) {
    const float*         Q    = (const float*)d_in[0];
    const float*         K    = (const float*)d_in[1];
    const float*         V    = (const float*)d_in[2];
    const unsigned char* mask = (const unsigned char*)d_in[3]; // jnp.bool_ (1B)

    float* ctx = (float*)d_out;                                  // [B,H,S,DK]
    float* wts = ctx + (size_t)2 * 16 * S_LEN * DK;              // [B,H,S,S]

    const size_t smem = (size_t)SMEM_FLOATS * sizeof(float);     // ~181 KB
    dim3 grid(2 * 16 * (S_LEN / QT));   // 4096 blocks
    sdpa_wmma_kernel<<<grid, 128, smem, stream>>>(Q, K, V, mask, ctx, wts);
}
